// TransformerNMT_44805098832333
// MI455X (gfx1250) — matrix operand order repaired and checked by reading, not compiled, not run on hardware
//
#include <hip/hip_runtime.h>
#include <hip/hip_bf16.h>
#include <math.h>

typedef __attribute__((ext_vector_type(16))) _Float16 f16x16;
typedef __attribute__((ext_vector_type(4)))  _Float16 f16x4;
typedef __attribute__((ext_vector_type(8)))  float    f32x8;
typedef __attribute__((ext_vector_type(4)))  float    f32x4;

#define TPB 256

// model dims
#define B_   8
#define S_   512
#define T_   256
#define D_   512
#define H_   8
#define DH_  64
#define DFF_ 2048
#define L_   6
#define NB_  32
#define VT_  32000

// ---------------- embedding gather (x = emb[ids] * sqrt(D)) ----------------
__global__ void embed_k(const int* __restrict__ ids, const float* __restrict__ emb,
                        float* __restrict__ out, int ntok, float scale) {
  long long i = (long long)blockIdx.x * blockDim.x + threadIdx.x;
  long long total = (long long)ntok * D_;
  if (i >= total) return;
  int tok = (int)(i / D_);
  int d   = (int)(i % D_);
  out[i] = emb[(long long)ids[tok] * D_ + d] * scale;
}

// ---------------- RMSNorm: one block per row of D=512 ----------------
__global__ void rmsnorm_k(const float* __restrict__ x, const float* __restrict__ w,
                          float* __restrict__ out) {
  long long row = blockIdx.x;
  const float* xr = x + row * D_;
  float* orow = out + row * D_;
  __shared__ float red[TPB];
  int tid = threadIdx.x;
  float s = 0.f;
  for (int i = tid; i < D_; i += TPB) { float v = xr[i]; s += v * v; }
  red[tid] = s; __syncthreads();
  for (int st = TPB / 2; st > 0; st >>= 1) {
    if (tid < st) red[tid] += red[tid + st];
    __syncthreads();
  }
  float inv = rsqrtf(red[0] / (float)D_ + 1e-8f);
  __syncthreads();
  for (int i = tid; i < D_; i += TPB) orow[i] = xr[i] * inv * w[i];
}

// ---------------- T5 relative position bias -> (H, qlen, klen) ----------------
__global__ void relbias_k(const float* __restrict__ emb, float* __restrict__ out,
                          int qlen, int klen) {
  int idx = blockIdx.x * blockDim.x + threadIdx.x;
  if (idx >= qlen * klen) return;
  int q = idx / klen, k = idx % klen;
  int n  = q - k;
  int sg = (n < 0) ? 1 : 0;
  int na = n < 0 ? -n : n;
  int bkt;
  if (na < NB_ / 2) {
    bkt = na;
  } else {
    float t = logf((float)na / (float)(NB_ / 2) + 1e-6f) /
              logf(128.0f / (float)(NB_ / 2)) * (float)(NB_ / 2 - 1);
    int vl = NB_ / 2 + (int)t;           // truncation toward zero == astype(int32)
    bkt = vl < (NB_ / 2 - 1) ? vl : (NB_ / 2 - 1);
  }
  bkt += sg * (NB_ / 2);
  bkt = bkt < 0 ? 0 : (bkt > NB_ - 1 ? NB_ - 1 : bkt);
  for (int h = 0; h < H_; ++h)
    out[((long long)h * qlen + q) * klen + k] = emb[bkt * H_ + h];
}

// ---------------- masked softmax over one (h,q) row of scores ----------------
__global__ void softmax_k(float* __restrict__ sc, const float* __restrict__ rel,
                          const int* __restrict__ kv_ids, int qlen, int klen,
                          int causal, float scale) {
  int q = blockIdx.x, h = blockIdx.y, tid = threadIdx.x;
  float* row = sc + ((long long)h * qlen + q) * klen;
  const float* rr = rel ? rel + ((long long)h * qlen + q) * klen : nullptr;
  __shared__ float red[TPB];
  float mx = -INFINITY;
  for (int k = tid; k < klen; k += TPB) {
    bool valid = (kv_ids[k] != 0) && (!causal || k <= q);
    float v = valid ? (row[k] * scale + (rr ? rr[k] : 0.f)) : -INFINITY;
    mx = fmaxf(mx, v);
  }
  red[tid] = mx; __syncthreads();
  for (int st = TPB / 2; st > 0; st >>= 1) {
    if (tid < st) red[tid] = fmaxf(red[tid], red[tid + st]);
    __syncthreads();
  }
  mx = red[0]; __syncthreads();
  float s = 0.f;
  for (int k = tid; k < klen; k += TPB) {
    bool valid = (kv_ids[k] != 0) && (!causal || k <= q);
    float e = 0.f;
    if (valid) e = __expf(row[k] * scale + (rr ? rr[k] : 0.f) - mx);
    row[k] = e; s += e;
  }
  red[tid] = s; __syncthreads();
  for (int st = TPB / 2; st > 0; st >>= 1) {
    if (tid < st) red[tid] += red[tid + st];
    __syncthreads();
  }
  float inv = red[0] > 0.f ? 1.f / red[0] : 0.f;
  for (int k = tid; k < klen; k += TPB) row[k] *= inv;
}

// ================= WMMA GEMM =================
// All launches guarantee M%128==0, N%64==0, K%32==0 (true for every GEMM in
// this model), so the kernel carries no bounds checks.
//
// LDS tiles are stored FRAGMENT-MAJOR: per 16x16(x32) subtile, lane L's 16
// halves are contiguous, so a fragment load is one 32B v16h load
// (2x ds_load_b128) with zero shuffle VALU.
//   A (16x32, ISA 7.12.2 table): element (m,k) -> lane ((k>>3)&1)*16 + m,
//     slot (k>>4)*8 + (k&7)          (4 consecutive k = 4 consecutive slots)
//   B (32x16): element (k,n) -> lane ((k>>3)&1)*16 + n, slot (k>>4)*8 + (k&7)
//
// Staging is software-pipelined: global loads for tile kt+1 are issued into
// registers BEFORE the WMMA block for tile kt, and converted/stored to the
// alternate LDS buffer AFTER it, so loads stay in flight under the WMMAs.
constexpr int BM = 128, BN = 64, BK = 32;
constexpr int ASZ = (BM / 16) * 512;   // 8 subtiles * 32 lanes * 16 halves
constexpr int BSZ = (BN / 16) * 512;   // 4 subtiles

template <bool BT>
static __device__ __forceinline__ void stage_load(
    const float* __restrict__ A, const float* __restrict__ Bm,
    int row0, int col0, int k0, int lda, int ldb, int tid,
    f32x4* __restrict__ aR, f32x4* __restrict__ bR) {
#pragma unroll
  for (int it = 0; it < 4; ++it) {
    int q = tid + it * TPB;            // 0..1023 quads of the 128x32 A tile
    int r = q >> 3;
    int c = (q & 7) * 4;
    aR[it] = *(const f32x4*)&A[(long long)(row0 + r) * lda + (k0 + c)];
  }
#pragma unroll
  for (int it = 0; it < 2; ++it) {
    int q = tid + it * TPB;            // 0..511 quads of the 32x64 B tile
    if (BT) {
      int k = (q & 7) * 4, n = q >> 3; // contiguous along K in memory
      bR[it] = *(const f32x4*)&Bm[(long long)(col0 + n) * ldb + (k0 + k)];
    } else {
      int n = q & 63, k = (q >> 6) * 4; // coalesced along N across lanes
#pragma unroll
      for (int e = 0; e < 4; ++e)
        bR[it][e] = Bm[(long long)(k0 + k + e) * ldb + (col0 + n)];
    }
  }
}

template <bool BT>
static __device__ __forceinline__ void stage_store(
    _Float16* __restrict__ As, _Float16* __restrict__ Bs, int tid,
    const f32x4* __restrict__ aR, const f32x4* __restrict__ bR) {
#pragma unroll
  for (int it = 0; it < 4; ++it) {
    int q = tid + it * TPB;
    int r = q >> 3;
    int c = (q & 7) * 4;
    f16x4 vh;
#pragma unroll
    for (int e = 0; e < 4; ++e) vh[e] = (_Float16)aR[it][e];
    int i    = r >> 4;
    int lane = ((c >> 3) & 1) * 16 + (r & 15);
    int e0   = ((c >> 4) << 3) + (c & 7);
    *(f16x4*)&As[i * 512 + lane * 16 + e0] = vh;
  }
#pragma unroll
  for (int it = 0; it < 2; ++it) {
    int q = tid + it * TPB;
    int n, k;
    if (BT) { k = (q & 7) * 4; n = q >> 3; }
    else    { n = q & 63;      k = (q >> 6) * 4; }
    f16x4 vh;
#pragma unroll
    for (int e = 0; e < 4; ++e) vh[e] = (_Float16)bR[it][e];
    int j    = n >> 4;
    int lane = ((k >> 3) & 1) * 16 + (n & 15);
    *(f16x4*)&Bs[j * 512 + lane * 16 + ((k >> 4) << 3) + (k & 7)] = vh;
  }
}

template <bool BT>
__launch_bounds__(TPB)
__global__ void gemm_wmma_k(const float* __restrict__ A, const float* __restrict__ Bm,
                            float* __restrict__ C,
                            const float* __restrict__ bias, const float* __restrict__ resid,
                            int M, int N, int K, int lda, int ldb, int ldc,
                            long long sA, long long sB, long long sC, int relu) {
  A  += (long long)blockIdx.z * sA;
  Bm += (long long)blockIdx.z * sB;
  C  += (long long)blockIdx.z * sC;
  if (resid) resid += (long long)blockIdx.z * sC;

  __shared__ __align__(16) _Float16 As[2][ASZ];
  __shared__ __align__(16) _Float16 Bs[2][BSZ];

  const int tid  = threadIdx.x;
  const int lane = tid & 31, wave = tid >> 5;
  const int ai0  = (wave >> 1) * 2;    // A subtile base (4 wave-rows)
  const int bj0  = (wave & 1) * 2;     // B subtile base (2 wave-cols)
  const int row0 = blockIdx.y * BM, col0 = blockIdx.x * BN;

  f32x8 acc[2][2] = {};
  f32x4 aR[4], bR[2];

  const int nk = K >> 5;
  stage_load<BT>(A, Bm, row0, col0, 0, lda, ldb, tid, aR, bR);
  stage_store<BT>(As[0], Bs[0], tid, aR, bR);
  __syncthreads();

  int buf = 0;
  for (int kt = 0; kt < nk; ++kt) {
    const bool more = (kt + 1 < nk);
    if (more)   // issue next tile's global loads; they fly under the WMMAs
      stage_load<BT>(A, Bm, row0, col0, (kt + 1) * BK, lda, ldb, tid, aR, bR);
    if ((kt + 2) * BK < K)   // L2 prefetch two tiles ahead
      __builtin_prefetch(&A[(long long)(row0 + (tid & (BM - 1))) * lda + (kt + 2) * BK], 0, 0);

    f16x16 af[2], bf[2];
#pragma unroll
    for (int i = 0; i < 2; ++i)
      af[i] = *(const f16x16*)&As[buf][(ai0 + i) * 512 + lane * 16];
#pragma unroll
    for (int j = 0; j < 2; ++j)
      bf[j] = *(const f16x16*)&Bs[buf][(bj0 + j) * 512 + lane * 16];
#pragma unroll
    for (int i = 0; i < 2; ++i)
#pragma unroll
      for (int j = 0; j < 2; ++j)
        acc[i][j] = __builtin_amdgcn_wmma_f32_16x16x32_f16(
            false, af[i], false, bf[j], (short)0, acc[i][j], false, false);

    if (more)   // convert + store the in-flight tile to the alternate buffer
      stage_store<BT>(As[buf ^ 1], Bs[buf ^ 1], tid, aR, bR);
    __syncthreads();
    buf ^= 1;
  }

  // epilogue: C VGPR e -> M = e + (lane>>4)*8, N = lane&15
  const int erow = (lane >> 4) * 8;
  const int ecol = lane & 15;
#pragma unroll
  for (int i = 0; i < 2; ++i) {
#pragma unroll
    for (int j = 0; j < 2; ++j) {
#pragma unroll
      for (int e = 0; e < 8; ++e) {
        int r = row0 + ai0 * 16 + i * 16 + erow + e;
        int c = col0 + bj0 * 16 + j * 16 + ecol;
        float v = acc[i][j][e];
        if (bias)  v += bias[c];
        if (relu)  v = v > 0.f ? v : 0.f;
        if (resid) v += resid[(long long)r * ldc + c];
        C[(long long)r * ldc + c] = v;
      }
    }
  }
}

// ---------------- host-side launch helper ----------------
// flags: bit0 = B transposed (N x K row-major), bit1 = ReLU
static void gemm(hipStream_t st, const float* A, const float* B, float* C,
                 const float* bias, const float* resid,
                 int M, int N, int K, int lda, int ldb, int ldc,
                 long long sA, long long sB, long long sC, int nb, int flags) {
  dim3 g(N / BN, M / BM, nb);
  int relu = (flags >> 1) & 1;
  if (flags & 1)
    gemm_wmma_k<true><<<g, TPB, 0, st>>>(A, B, C, bias, resid, M, N, K,
                                         lda, ldb, ldc, sA, sB, sC, relu);
  else
    gemm_wmma_k<false><<<g, TPB, 0, st>>>(A, B, C, bias, resid, M, N, K,
                                          lda, ldb, ldc, sA, sB, sC, relu);
}

extern "C" void kernel_launch(void* const* d_in, const int* in_sizes, int n_in,
                              void* d_out, int out_size, void* d_ws, size_t ws_size,
                              hipStream_t stream) {
  (void)in_sizes; (void)n_in; (void)out_size; (void)ws_size;
  const int*   src_ids   = (const int*)d_in[0];
  const int*   tgt_ids   = (const int*)d_in[1];
  const float* src_emb   = (const float*)d_in[2];
  const float* tgt_emb   = (const float*)d_in[3];
  const float* rel_enc   = (const float*)d_in[4];
  const float* rel_dec   = (const float*)d_in[5];
  const float* enc_norm1 = (const float*)d_in[6];
  const float* enc_norm2 = (const float*)d_in[7];
  const float* enc_Wq    = (const float*)d_in[8];
  const float* enc_Wk    = (const float*)d_in[9];
  const float* enc_Wv    = (const float*)d_in[10];
  const float* enc_Wo    = (const float*)d_in[11];
  const float* enc_W1    = (const float*)d_in[12];
  const float* enc_b1    = (const float*)d_in[13];
  const float* enc_W2    = (const float*)d_in[14];
  const float* enc_b2    = (const float*)d_in[15];
  const float* dec_norm1 = (const float*)d_in[16];
  const float* dec_norm2 = (const float*)d_in[17];
  const float* dec_norm3 = (const float*)d_in[18];
  const float* dec_sWq   = (const float*)d_in[19];
  const float* dec_sWk   = (const float*)d_in[20];
  const float* dec_sWv   = (const float*)d_in[21];
  const float* dec_sWo   = (const float*)d_in[22];
  const float* dec_cWq   = (const float*)d_in[23];
  const float* dec_cWk   = (const float*)d_in[24];
  const float* dec_cWv   = (const float*)d_in[25];
  const float* dec_cWo   = (const float*)d_in[26];
  const float* dec_W1    = (const float*)d_in[27];
  const float* dec_b1    = (const float*)d_in[28];
  const float* dec_W2    = (const float*)d_in[29];
  const float* dec_b2    = (const float*)d_in[30];
  const float* final_w   = (const float*)d_in[31];
  float* out = (float*)d_out;

  // workspace layout (fp32 elements)
  float* ws = (float*)d_ws;
  const long long BSD = (long long)B_ * S_ * D_;
  const long long BTD = (long long)B_ * T_ * D_;
  float* x  = ws;              // B*S*D  (encoder stream / memory)
  float* h  = x  + BSD;        // B*S*D  (normed activations)
  float* qb = h  + BSD;        // B*S*D
  float* kb = qb + BSD;        // B*S*D
  float* vb = kb + BSD;        // B*S*D
  float* ab = vb + BSD;        // B*S*D  (attention output pre-Wo)
  float* ff = ab + BSD;        // B*S*DFF
  float* sc = ff + (long long)B_ * S_ * DFF_;      // H*S*S (per-batch scores)
  float* rE = sc + (long long)H_ * S_ * S_;        // H*S*S rel bias enc
  float* rD = rE + (long long)H_ * S_ * S_;        // H*T*T rel bias dec
  float* y  = rD + (long long)H_ * T_ * T_;        // B*T*D (decoder stream)

  const int BS = B_ * S_, BT = B_ * T_;
  const float embscale = sqrtf((float)D_);

  relbias_k<<<(S_ * S_ + TPB - 1) / TPB, TPB, 0, stream>>>(rel_enc, rE, S_, S_);
  relbias_k<<<(T_ * T_ + TPB - 1) / TPB, TPB, 0, stream>>>(rel_dec, rD, T_, T_);

  // ================= encoder =================
  embed_k<<<(int)((BSD + TPB - 1) / TPB), TPB, 0, stream>>>(src_ids, src_emb, x, BS, embscale);
  for (int l = 0; l < L_; ++l) {
    const float* Wq = enc_Wq + (long long)l * D_ * D_;
    const float* Wk = enc_Wk + (long long)l * D_ * D_;
    const float* Wv = enc_Wv + (long long)l * D_ * D_;
    const float* Wo = enc_Wo + (long long)l * D_ * D_;
    rmsnorm_k<<<BS, TPB, 0, stream>>>(x, enc_norm1 + l * D_, h);
    gemm(stream, h, Wq, qb, nullptr, nullptr, BS, D_, D_, D_, D_, D_, 0, 0, 0, 1, 0);
    gemm(stream, h, Wk, kb, nullptr, nullptr, BS, D_, D_, D_, D_, D_, 0, 0, 0, 1, 0);
    gemm(stream, h, Wv, vb, nullptr, nullptr, BS, D_, D_, D_, D_, D_, 0, 0, 0, 1, 0);
    for (int b = 0; b < B_; ++b) {
      const long long ob = (long long)b * S_ * D_;
      gemm(stream, qb + ob, kb + ob, sc, nullptr, nullptr,
           S_, S_, DH_, D_, D_, S_, DH_, DH_, (long long)S_ * S_, H_, 1);
      softmax_k<<<dim3(S_, H_), TPB, 0, stream>>>(sc, rE, src_ids + b * S_, S_, S_, 0, 0.125f);
      gemm(stream, sc, vb + ob, ab + ob, nullptr, nullptr,
           S_, DH_, S_, S_, D_, D_, (long long)S_ * S_, DH_, DH_, H_, 0);
    }
    gemm(stream, ab, Wo, x, nullptr, x, BS, D_, D_, D_, D_, D_, 0, 0, 0, 1, 0);
    rmsnorm_k<<<BS, TPB, 0, stream>>>(x, enc_norm2 + l * D_, h);
    gemm(stream, h, enc_W1 + (long long)l * D_ * DFF_, ff, enc_b1 + l * DFF_, nullptr,
         BS, DFF_, D_, D_, DFF_, DFF_, 0, 0, 0, 1, 2);
    gemm(stream, ff, enc_W2 + (long long)l * DFF_ * D_, x, enc_b2 + l * D_, x,
         BS, D_, DFF_, DFF_, D_, D_, 0, 0, 0, 1, 0);
  }
  // x is now encoder memory

  // ================= decoder =================
  embed_k<<<(int)((BTD + TPB - 1) / TPB), TPB, 0, stream>>>(tgt_ids, tgt_emb, y, BT, embscale);
  for (int l = 0; l < L_; ++l) {
    const long long lDD = (long long)l * D_ * D_;
    // ---- masked self-attention ----
    rmsnorm_k<<<BT, TPB, 0, stream>>>(y, dec_norm1 + l * D_, h);
    gemm(stream, h, dec_sWq + lDD, qb, nullptr, nullptr, BT, D_, D_, D_, D_, D_, 0, 0, 0, 1, 0);
    gemm(stream, h, dec_sWk + lDD, kb, nullptr, nullptr, BT, D_, D_, D_, D_, D_, 0, 0, 0, 1, 0);
    gemm(stream, h, dec_sWv + lDD, vb, nullptr, nullptr, BT, D_, D_, D_, D_, D_, 0, 0, 0, 1, 0);
    for (int b = 0; b < B_; ++b) {
      const long long ob = (long long)b * T_ * D_;
      gemm(stream, qb + ob, kb + ob, sc, nullptr, nullptr,
           T_, T_, DH_, D_, D_, T_, DH_, DH_, (long long)T_ * T_, H_, 1);
      softmax_k<<<dim3(T_, H_), TPB, 0, stream>>>(sc, rD, tgt_ids + b * T_, T_, T_, 1, 0.125f);
      gemm(stream, sc, vb + ob, ab + ob, nullptr, nullptr,
           T_, DH_, T_, T_, D_, D_, (long long)T_ * T_, DH_, DH_, H_, 0);
    }
    gemm(stream, ab, dec_sWo + lDD, y, nullptr, y, BT, D_, D_, D_, D_, D_, 0, 0, 0, 1, 0);
    // ---- cross-attention (K/V from un-normed encoder memory) ----
    rmsnorm_k<<<BT, TPB, 0, stream>>>(y, dec_norm2 + l * D_, h);
    gemm(stream, h, dec_cWq + lDD, qb, nullptr, nullptr, BT, D_, D_, D_, D_, D_, 0, 0, 0, 1, 0);
    gemm(stream, x, dec_cWk + lDD, kb, nullptr, nullptr, BS, D_, D_, D_, D_, D_, 0, 0, 0, 1, 0);
    gemm(stream, x, dec_cWv + lDD, vb, nullptr, nullptr, BS, D_, D_, D_, D_, D_, 0, 0, 0, 1, 0);
    for (int b = 0; b < B_; ++b) {
      const long long oq  = (long long)b * T_ * D_;
      const long long okv = (long long)b * S_ * D_;
      gemm(stream, qb + oq, kb + okv, sc, nullptr, nullptr,
           T_, S_, DH_, D_, D_, S_, DH_, DH_, (long long)T_ * S_, H_, 1);
      softmax_k<<<dim3(T_, H_), TPB, 0, stream>>>(sc, (const float*)nullptr,
                                                  src_ids + b * S_, T_, S_, 0, 0.125f);
      gemm(stream, sc, vb + okv, ab + oq, nullptr, nullptr,
           T_, DH_, S_, S_, D_, D_, (long long)T_ * S_, DH_, DH_, H_, 0);
    }
    gemm(stream, ab, dec_cWo + lDD, y, nullptr, y, BT, D_, D_, D_, D_, D_, 0, 0, 0, 1, 0);
    // ---- FFN ----
    rmsnorm_k<<<BT, TPB, 0, stream>>>(y, dec_norm3 + l * D_, h);
    gemm(stream, h, dec_W1 + (long long)l * D_ * DFF_, ff, dec_b1 + l * DFF_, nullptr,
         BT, DFF_, D_, D_, DFF_, DFF_, 0, 0, 0, 1, 2);
    gemm(stream, ff, dec_W2 + (long long)l * DFF_ * D_, y, dec_b2 + l * D_, y,
         BT, D_, DFF_, DFF_, D_, D_, 0, 0, 0, 1, 0);
  }

  // final norm + tied lm head: logits = rmsnorm(y) @ tgt_emb^T
  rmsnorm_k<<<BT, TPB, 0, stream>>>(y, final_w, h);
  gemm(stream, h, tgt_emb, out, nullptr, nullptr, BT, VT_, D_, D_, D_, VT_, 0, 0, 0, 1, 1);
}
